// CompactCrossAttention_40793599378138
// MI455X (gfx1250) — matrix operand order repaired and checked by reading, not compiled, not run on hardware
//
#include <hip/hip_runtime.h>
#include <hip/hip_bf16.h>

// ---------------------------------------------------------------------------
// CompactCrossAttention on gfx1250 (MI455X): bf16 WMMA pipeline + TDM staging.
//   B=2, QL=1024, KL=4096, H=1024, NH=16, HD=64
// ---------------------------------------------------------------------------

typedef __attribute__((ext_vector_type(16))) __bf16        v16bf;
typedef __attribute__((ext_vector_type(8)))  float         v8f;
typedef __attribute__((ext_vector_type(4)))  unsigned int  u32x4;
typedef __attribute__((ext_vector_type(4)))  int           i32x4;
typedef __attribute__((ext_vector_type(8)))  int           i32x8;

#define B_   2
#define QL_  1024
#define KL_  4096
#define H_   1024
#define NH_  16
#define HD_  64

__device__ __forceinline__ unsigned short f2bf(float f) {
    unsigned int u = __float_as_uint(f);
    unsigned int r = u + 0x7FFFu + ((u >> 16) & 1u);   // round-to-nearest-even
    return (unsigned short)(r >> 16);
}

// A-fragment k order used here (contiguous, not AMD's table order): lane L holds row m=L%16, dword v holds
// K = {off, off+1} with off = (L/16)*16 + v*2.
__device__ __forceinline__ int a_k_off(int v, int hgrp) {
    return hgrp * 16 + v * 2;
}

union AFrag { v16bf v; unsigned int u[8]; };

// ---- DPP16 row (16-lane) rotate-reductions: pure VALU, no LDS traffic -----
template <int CTRL>
__device__ __forceinline__ float dpp_rorf(float x) {
    return __int_as_float(__builtin_amdgcn_update_dpp(
        0, __float_as_int(x), CTRL, 0xF, 0xF, true));
}
__device__ __forceinline__ float rowmax16(float x) {   // row_ror:1,2,4,8
    x = fmaxf(x, dpp_rorf<0x121>(x));
    x = fmaxf(x, dpp_rorf<0x122>(x));
    x = fmaxf(x, dpp_rorf<0x124>(x));
    x = fmaxf(x, dpp_rorf<0x128>(x));
    return x;
}
__device__ __forceinline__ float rowsum16(float x) {
    x += dpp_rorf<0x121>(x);
    x += dpp_rorf<0x122>(x);
    x += dpp_rorf<0x124>(x);
    x += dpp_rorf<0x128>(x);
    return x;
}

// ---- Tensor Data Mover: 2D tile load (arity hedged across toolchains) -----
__device__ __forceinline__ void tdm_load_2d(u32x4 g0, i32x8 g1) {
    i32x4 z4 = {};
#if defined(__clang_major__) && (__clang_major__ >= 23)
    i32x8 z8 = {};
    __builtin_amdgcn_tensor_load_to_lds(g0, g1, z4, z4, z8, 0);
#else
    __builtin_amdgcn_tensor_load_to_lds(g0, g1, z4, z4, 0);
#endif
}

// ---------------------------------------------------------------------------
// fp32 -> bf16 conversion
// ---------------------------------------------------------------------------
__global__ __launch_bounds__(256) void f32_to_bf16_kernel(const float* __restrict__ in,
                                                          unsigned short* __restrict__ out,
                                                          int n) {
    int i = blockIdx.x * 256 + threadIdx.x;
    if (i < n) out[i] = f2bf(in[i]);
}

// ---------------------------------------------------------------------------
// bf16 GEMM: C[M,N] = A[M,K] @ Bm[K,N]   (row-major bf16, f32 accumulate)
// WG = 256 threads (8 waves). WG tile 128x128; wave tile 32x64 (2x4 accums).
// A tile (128x32) staged via TENSOR_LOAD_TO_LDS (TDM, wave 0); B tile (32x128)
// staged transposed via b128 loads + ds stores.
// EPI 0: f32 row-major out.  EPI 1: Q scatter [B,NH,QL,HD] bf16, pre-scaled
// by 1/sqrt(HD).  EPI 2: KV scatter -> Kb [B,NH,KL,HD] + Vt [B,NH,HD,KL].
// ---------------------------------------------------------------------------
template <int EPI>
__global__ __launch_bounds__(256) void gemm_bf16_kernel(
        const unsigned short* __restrict__ A, const unsigned short* __restrict__ Bm,
        int M, int N, int K,
        unsigned short* __restrict__ outA, unsigned short* __restrict__ outB,
        float* __restrict__ outF) {
    __shared__ unsigned short ldsA[128 * 32];  // [m][k]   (TDM destination)
    __shared__ unsigned short ldsB[128 * 32];  // [n][k]   (transposed)

    const int tid  = threadIdx.x;
    const int lane = tid & 31;
    const int wave = tid >> 5;
    const int wm   = wave & 3;     // 4 waves along M
    const int wn   = wave >> 2;    // 2 waves along N
    const int m0   = blockIdx.x * 128;
    const int n0   = blockIdx.y * 128;
    const int nlan = lane & 15;
    const int hgrp = lane >> 4;

    v8f acc[2][4];
#pragma unroll
    for (int i = 0; i < 2; ++i)
#pragma unroll
        for (int j = 0; j < 4; ++j) acc[i][j] = {};

    const unsigned lds_a_base = (unsigned)(uintptr_t)(&ldsA[0]);

    for (int kt = 0; kt < K; kt += 32) {
        // --- A tile via Tensor Data Mover (one descriptor per WG) ---
        if (wave == 0) {
            unsigned long long ga =
                (unsigned long long)(uintptr_t)(A + (size_t)m0 * K + kt);
            u32x4 g0;
            g0.x = 1u;                                   // count=1, user mode
            g0.y = lds_a_base;                           // lds_addr
            g0.z = (unsigned)ga;                         // global_addr[31:0]
            g0.w = (unsigned)((ga >> 32) & 0x01FFFFFFull) | 0x80000000u; // [56:32] | type=2
            i32x8 g1;
            g1.s0 = 0x00010000;                          // data_size = 2 bytes
            g1.s1 = (K & 0xFFFF) << 16;                  // tensor_dim0 lo16
            g1.s2 = (K >> 16) | ((M & 0xFFFF) << 16);    // td0 hi16 | tensor_dim1 lo16
            g1.s3 = (M >> 16) | (32 << 16);              // td1 hi16 | tile_dim0 = 32
            g1.s4 = 128;                                 // tile_dim1 = 128, tile_dim2 = 0
            g1.s5 = K;                                   // tensor_dim0_stride lo32
            g1.s6 = 0;
            g1.s7 = 0;
            tdm_load_2d(g0, g1);
            __builtin_amdgcn_s_wait_tensorcnt(0);
        }
        // --- B tile 32x128, store transposed into ldsB[n][k] ---
        {
            int k = tid >> 3, g = tid & 7;
#pragma unroll
            for (int rep = 0; rep < 2; ++rep) {
                int ng = g + rep * 8;
                union { u32x4 q; unsigned short h[8]; } t;
                t.q = *(const u32x4*)(Bm + (size_t)(kt + k) * N + n0 + ng * 8);
#pragma unroll
                for (int j = 0; j < 8; ++j) ldsB[(ng * 8 + j) * 32 + k] = t.h[j];
            }
        }
        __syncthreads();

        AFrag af[2], bfr[4];
#pragma unroll
        for (int mi = 0; mi < 2; ++mi) {
            int m = wm * 32 + mi * 16 + nlan;
#pragma unroll
            for (int v = 0; v < 8; ++v)
                af[mi].u[v] = *(const unsigned int*)(ldsA + m * 32 + a_k_off(v, hgrp));
        }
#pragma unroll
        for (int ni = 0; ni < 4; ++ni) {
            int n = wn * 64 + ni * 16 + nlan;
#pragma unroll
            for (int v = 0; v < 8; ++v)
                bfr[ni].u[v] = *(const unsigned int*)(ldsB + n * 32 + hgrp * 16 + v * 2);
        }
#pragma unroll
        for (int mi = 0; mi < 2; ++mi)
#pragma unroll
            for (int ni = 0; ni < 4; ++ni)
                acc[mi][ni] = __builtin_amdgcn_wmma_f32_16x16x32_bf16(
                    false, af[mi].v, false, bfr[ni].v, (short)0, acc[mi][ni], false, false);
        __syncthreads();
    }

    // Epilogue: C layout -> lane holds (m = v + 8*(lane/16), n = lane%16)
#pragma unroll
    for (int mi = 0; mi < 2; ++mi)
#pragma unroll
        for (int ni = 0; ni < 4; ++ni)
#pragma unroll
            for (int v = 0; v < 8; ++v) {
                int gm = m0 + wm * 32 + mi * 16 + v + 8 * hgrp;
                int gn = n0 + wn * 64 + ni * 16 + nlan;
                float val = acc[mi][ni][v];
                if constexpr (EPI == 0) {
                    outF[(size_t)gm * N + gn] = val;
                } else if constexpr (EPI == 1) {
                    int b = gm >> 10, q = gm & (QL_ - 1);
                    int h = gn >> 6, d = gn & 63;
                    outA[(((size_t)(b * NH_ + h)) * QL_ + q) * HD_ + d] =
                        f2bf(val * 0.125f);   // fold 1/sqrt(HD) into Q
                } else {
                    int b = gm >> 12, kl = gm & (KL_ - 1);
                    if (gn < H_) {
                        int h = gn >> 6, d = gn & 63;
                        outA[(((size_t)(b * NH_ + h)) * KL_ + kl) * HD_ + d] = f2bf(val);
                    } else {
                        int c = gn - H_;
                        int h = c >> 6, d = c & 63;
                        outB[(((size_t)(b * NH_ + h)) * HD_ + d) * KL_ + kl] = f2bf(val);
                    }
                }
            }
}

// ---------------------------------------------------------------------------
// Flash attention: grid (B*NH, QL/64), block 128 (4 waves, 16 q-rows each).
// Qb [B,NH,QL,HD] bf16 (pre-scaled), Kb [B,NH,KL,HD] bf16, Vt [B,NH,HD,KL].
// Out: Ab [B*QL, H] bf16 (head-interleaved row-major, ready for out-proj).
// ---------------------------------------------------------------------------
__global__ __launch_bounds__(128) void flash_attn_kernel(
        const unsigned short* __restrict__ Qb, const unsigned short* __restrict__ Kb,
        const unsigned short* __restrict__ Vt, unsigned short* __restrict__ Ab) {
    __shared__ unsigned short ldsP[4 * 16 * 32];   // per-wave 16x32 bf16 P tile

    const int tid  = threadIdx.x;
    const int lane = tid & 31;
    const int wave = tid >> 5;
    const int nlan = lane & 15;
    const int hgrp = lane >> 4;
    const int bh   = blockIdx.x;                   // b*NH + h
    const int q0   = blockIdx.y * 64 + wave * 16;

    const unsigned short* Qp = Qb + ((size_t)bh * QL_ + q0) * HD_;
    const unsigned short* Kp = Kb + (size_t)bh * KL_ * HD_;
    const unsigned short* Vp = Vt + (size_t)bh * HD_ * KL_;
    unsigned short* ldsW = ldsP + wave * 16 * 32;

    // Q A-fragments, resident for the whole KL loop (d = 0..31, 32..63)
    AFrag qf[2];
#pragma unroll
    for (int kk = 0; kk < 2; ++kk)
#pragma unroll
        for (int v = 0; v < 8; ++v)
            qf[kk].u[v] = *(const unsigned int*)(Qp + (size_t)nlan * HD_ + kk * 32 + a_k_off(v, hgrp));

    v8f o[4];
#pragma unroll
    for (int t = 0; t < 4; ++t) o[t] = {};
    float mrow[8], lrow[8];
#pragma unroll
    for (int v = 0; v < 8; ++v) { mrow[v] = -3.0e38f; lrow[v] = 0.0f; }

    for (int kl0 = 0; kl0 < KL_; kl0 += 32) {
        if (kl0 + 32 < KL_) {
            __builtin_prefetch(Kp + (size_t)(kl0 + 32 + lane) * HD_, 0, 0);
            __builtin_prefetch(Vp + (size_t)(lane << 1) * KL_ + kl0 + 32, 0, 0);
        }
        // S tiles: s[st] = Q(16x64) @ K^T over kl columns [kl0+16*st, +16)
        v8f s[2];
#pragma unroll
        for (int st = 0; st < 2; ++st) {
            s[st] = {};
            int klrow = kl0 + st * 16 + nlan;   // per-lane column index (B-layout n)
#pragma unroll
            for (int kk = 0; kk < 2; ++kk) {
                AFrag kf;
                const u32x4* src = (const u32x4*)(Kp + (size_t)klrow * HD_ + kk * 32 + hgrp * 16);
                u32x4 t0 = src[0], t1 = src[1];
                kf.u[0] = t0.x; kf.u[1] = t0.y; kf.u[2] = t0.z; kf.u[3] = t0.w;
                kf.u[4] = t1.x; kf.u[5] = t1.y; kf.u[6] = t1.z; kf.u[7] = t1.w;
                s[st] = __builtin_amdgcn_wmma_f32_16x16x32_bf16(
                    false, qf[kk].v, false, kf.v, (short)0, s[st], false, false);
            }
        }
        // Online softmax over the 32 new columns (DPP row reductions);
        // write P (bf16) to LDS for the layout transpose.
#pragma unroll
        for (int v = 0; v < 8; ++v) {
            float s0 = s[0][v], s1 = s[1][v];   // Q pre-scaled by 1/sqrt(HD)
            float t  = rowmax16(fmaxf(s0, s1));
            float nm   = fmaxf(mrow[v], t);
            float corr = __expf(mrow[v] - nm);
            float p0   = __expf(s0 - nm);
            float p1   = __expf(s1 - nm);
            float rs   = rowsum16(p0 + p1);
            lrow[v] = corr * lrow[v] + rs;
            mrow[v] = nm;
#pragma unroll
            for (int t4 = 0; t4 < 4; ++t4) o[t4][v] = o[t4][v] * corr;
            int m = v + 8 * hgrp;
            ldsW[m * 32 + nlan]      = f2bf(p0);
            ldsW[m * 32 + 16 + nlan] = f2bf(p1);
        }
        asm volatile("s_wait_dscnt 0x0" ::: "memory");   // wave-local LDS transpose sync

        AFrag pf;
#pragma unroll
        for (int v = 0; v < 8; ++v)
            pf.u[v] = *(const unsigned int*)(ldsW + nlan * 32 + a_k_off(v, hgrp));

        // O(16x64) += P(16x32) @ V(32x64); Vt is kl-contiguous -> b128 loads.
#pragma unroll
        for (int t4 = 0; t4 < 4; ++t4) {
            AFrag vf;
            int d = t4 * 16 + nlan;
            const u32x4* src = (const u32x4*)(Vp + (size_t)d * KL_ + kl0 + hgrp * 16);
            u32x4 t0 = src[0], t1 = src[1];
            vf.u[0] = t0.x; vf.u[1] = t0.y; vf.u[2] = t0.z; vf.u[3] = t0.w;
            vf.u[4] = t1.x; vf.u[5] = t1.y; vf.u[6] = t1.z; vf.u[7] = t1.w;
            o[t4] = __builtin_amdgcn_wmma_f32_16x16x32_bf16(
                false, pf.v, false, vf.v, (short)0, o[t4], false, false);
        }
    }

    // Normalize and scatter to Ab [B*QL, H] (col = h*64 + d)
    const int b = bh >> 4, h = bh & 15;
#pragma unroll
    for (int t4 = 0; t4 < 4; ++t4)
#pragma unroll
        for (int v = 0; v < 8; ++v) {
            int m = v + 8 * hgrp;
            float val = o[t4][v] / lrow[v];
            int row = b * QL_ + q0 + m;
            int col = h * HD_ + t4 * 16 + nlan;
            Ab[(size_t)row * H_ + col] = f2bf(val);
        }
}

// ---------------------------------------------------------------------------
// Launch
// ---------------------------------------------------------------------------
extern "C" void kernel_launch(void* const* d_in, const int* in_sizes, int n_in,
                              void* d_out, int out_size, void* d_ws, size_t ws_size,
                              hipStream_t stream) {
    (void)in_sizes; (void)n_in; (void)out_size; (void)ws_size;
    const float* query  = (const float*)d_in[0];   // [B,QL,H]
    const float* keyval = (const float*)d_in[1];   // [B,KL,H]
    const float* w_q    = (const float*)d_in[2];   // [H,H]
    const float* w_kv   = (const float*)d_in[3];   // [H,2H]
    const float* w_out  = (const float*)d_in[4];   // [H,H]
    float* out = (float*)d_out;                    // [B,QL,H] f32

    // Workspace layout (bf16 elements), total 34M elems = 68 MB
    unsigned short* ws   = (unsigned short*)d_ws;
    constexpr size_t N_QBF = (size_t)B_ * QL_ * H_;        // 2M
    constexpr size_t N_KVB = (size_t)B_ * KL_ * H_;        // 8M
    constexpr size_t N_WQ  = (size_t)H_ * H_;              // 1M
    constexpr size_t N_WKV = (size_t)H_ * 2 * H_;          // 2M
    constexpr size_t N_WO  = (size_t)H_ * H_;              // 1M
    constexpr size_t N_QB  = (size_t)B_ * NH_ * QL_ * HD_; // 2M
    constexpr size_t N_KB  = (size_t)B_ * NH_ * KL_ * HD_; // 8M
    constexpr size_t N_VT  = N_KB;                         // 8M
    unsigned short* qbf  = ws;
    unsigned short* kvbf = qbf + N_QBF;
    unsigned short* wqb  = kvbf + N_KVB;
    unsigned short* wkvb = wqb + N_WQ;
    unsigned short* wob  = wkvb + N_WKV;
    unsigned short* Qb   = wob + N_WO;
    unsigned short* Kb   = Qb + N_QB;
    unsigned short* Vt   = Kb + N_KB;
    unsigned short* Ab   = Vt + N_VT;

    auto cvt = [&](const float* src, unsigned short* dst, size_t n) {
        f32_to_bf16_kernel<<<dim3((unsigned)((n + 255) / 256)), dim3(256), 0, stream>>>(src, dst, (int)n);
    };
    cvt(query,  qbf,  N_QBF);
    cvt(keyval, kvbf, N_KVB);
    cvt(w_q,    wqb,  N_WQ);
    cvt(w_kv,   wkvb, N_WKV);
    cvt(w_out,  wob,  N_WO);

    // Q projection: [2048,1024] = qbf @ wqb, scatter (pre-scaled) to Qb
    gemm_bf16_kernel<1><<<dim3(2048 / 128, 1024 / 128), 256, 0, stream>>>(
        qbf, wqb, 2048, 1024, 1024, Qb, nullptr, nullptr);
    // KV projection: [8192,2048] = kvbf @ wkvb, scatter to Kb + transposed Vt
    gemm_bf16_kernel<2><<<dim3(8192 / 128, 2048 / 128), 256, 0, stream>>>(
        kvbf, wkvb, 8192, 2048, 1024, Kb, Vt, nullptr);
    // Attention (flash, online softmax)
    flash_attn_kernel<<<dim3(B_ * NH_, QL_ / 64), 128, 0, stream>>>(Qb, Kb, Vt, Ab);
    // Output projection: f32 result straight to d_out
    gemm_bf16_kernel<0><<<dim3(2048 / 128, 1024 / 128), 256, 0, stream>>>(
        Ab, wob, 2048, 1024, 1024, nullptr, nullptr, out);
}
